// WaveNet_62740882260080
// MI455X (gfx1250) — compile-verified, matrix operand order checked by reading, not run on hardware
//
#include <hip/hip_runtime.h>
#include <hip/hip_bf16.h>
#include <math.h>

// ---------------------------------------------------------------------------
// WaveNet forecaster for MI455X (gfx1250, wave32, WMMA + TDM).
//   Encoder: 23 layers of  z = W_l * [h_past ; h]  as (256 x 128) x (128 x B*T)
//            GEMM in v_wmma_f32_16x16x32_f16; W_l staged to LDS by the Tensor
//            Data Mover (tensor_load_to_lds + s_wait_tensorcnt).
//   Decoder: persistent single-workgroup kernel, 64 steps x 24 layers,
//            per-layer (256x128)x(128x32) WMMA GEMM + 256x1536x32 output GEMM.
// ---------------------------------------------------------------------------

typedef _Float16 v16h __attribute__((ext_vector_type(16)));
typedef _Float16 v8h  __attribute__((ext_vector_type(8)));
typedef float    v8f  __attribute__((ext_vector_type(8)));
typedef unsigned int v4u __attribute__((ext_vector_type(4)));
typedef int      v4i  __attribute__((ext_vector_type(4)));
typedef int      v8i  __attribute__((ext_vector_type(8)));

#define T_ENC 1024
#define BATCH 32
#define RCH   64
#define NLAY  24
#define QLEN  192     // ring: last 128 encoder steps + 64 decoder steps
#define DECL  64
#define HID   256
#define SKIPK (NLAY * RCH)       // 1536
#define WLAYER (16 * 4 * 32 * 16)  // 32768 f16 = 64 KB per layer, fragment order

#if defined(__has_builtin)
#  if __has_builtin(__builtin_amdgcn_tensor_load_to_lds) && \
      __has_builtin(__builtin_amdgcn_s_wait_tensorcnt)
#    define USE_TDM 1
#  endif
#endif
#ifndef USE_TDM
#  define USE_TDM 0
#endif

// ISA 7.12.2: 16-bit A (16x32): lanes 0-15 rows M, element j -> K offset
__device__ __forceinline__ int frag_k(int half, int j) {
    return (j < 8) ? (half * 8 + j) : (16 + half * 8 + (j - 8));
}

__device__ __forceinline__ float sigm(float x) { return 1.0f / (1.0f + __expf(-x)); }

#if USE_TDM
// TDM: DMA a contiguous block of n2b f16 elements (n2b <= 65535) into LDS.
// D# per ISA 8.3/8.4: group0 = {count=1 | lds_addr | global_addr | type=2},
// group1 = {data_size=2B, tensor_dim0=tile_dim0=n2b, dims1..=1, stride0=n2b}.
// This toolchain exposes the 6-arg builtin:
//   (v4u g0, v8i g1, v4i g2, v4i g3, v8i extra, i32 cpol)
__device__ __forceinline__ void tdm_load_block(const void* g, unsigned lds_addr,
                                               unsigned n2b) {
    unsigned long long ga = (unsigned long long)(uintptr_t)g;
    v4u g0;
    g0[0] = 1u;                                              // count=1 (valid user D#)
    g0[1] = lds_addr;                                        // LDS byte address
    g0[2] = (unsigned)ga;                                    // global_addr[31:0]
    g0[3] = (unsigned)((ga >> 32) & 0x01FFFFFFull) | (2u << 30);  // [56:32] | type=2
    v8i g1;
    g1[0] = (int)(1u << 16);                                 // data_size=1 -> 2 bytes
    g1[1] = (int)((n2b & 0xFFFFu) << 16);                    // tensor_dim0[15:0]
    g1[2] = (int)(((n2b >> 16) & 0xFFFFu) | (1u << 16));     // dim0[31:16] | tensor_dim1=1
    g1[3] = (int)((n2b & 0xFFFFu) << 16);                    // tile_dim0
    g1[4] = 1;                                               // tile_dim1=1
    g1[5] = (int)n2b;                                        // tensor_dim0_stride[31:0]
    g1[6] = 0;
    g1[7] = 0;
    v4i zz4 = {0, 0, 0, 0};
    v8i zz8 = {0, 0, 0, 0, 0, 0, 0, 0};
    __builtin_amdgcn_tensor_load_to_lds(g0, g1, zz4, zz4, zz8, 0);
    __builtin_amdgcn_s_wait_tensorcnt((short)0);
}
#endif

// ---------------------------------------------------------------------------
// Weight swizzle: cnn weights (L, 256, 64, 2 taps) -> f16 fragment order
// out[l][mt(16)][kt(4)][lane(32)][j(16)], K = kt*32 + frag_k, tap = K>=64.
// ---------------------------------------------------------------------------
__global__ __launch_bounds__(256) void swizzle_cnn_w_kernel(const float* __restrict__ w,
                                                            _Float16* __restrict__ out) {
    int gid = blockIdx.x * blockDim.x + threadIdx.x;
    const int total = NLAY * WLAYER;
    if (gid >= total) return;
    int j    = gid & 15;
    int lane = (gid >> 4) & 31;
    int kt   = (gid >> 9) & 3;
    int mt   = (gid >> 11) & 15;
    int l    = gid >> 15;
    int m  = mt * 16 + (lane & 15);
    int k  = kt * 32 + frag_k(lane >> 4, j);
    int tap = k >> 6;
    int ch  = k & 63;
    out[gid] = (_Float16)w[(((size_t)l * 256 + m) * 64 + ch) * 2 + tap];
}

// out1_w (256, 1536) -> fragments out[mt(16)][kt(48)][lane(32)][j(16)]
__global__ __launch_bounds__(256) void swizzle_out1_w_kernel(const float* __restrict__ w,
                                                             _Float16* __restrict__ out) {
    int gid = blockIdx.x * blockDim.x + threadIdx.x;
    const int total = 16 * 48 * 32 * 16;
    if (gid >= total) return;
    int j    = gid & 15;
    int lane = (gid >> 4) & 31;
    int rest = gid >> 9;
    int kt   = rest % 48;
    int mt   = rest / 48;
    int m = mt * 16 + (lane & 15);
    int k = kt * 32 + frag_k(lane >> 4, j);
    out[gid] = (_Float16)w[(size_t)m * SKIPK + k];
}

// ---------------------------------------------------------------------------
// Encoder init: h0 = tanh(w_h*x + b_h), c0 = tanh(w_c*x + b_c)  (SRC==1)
// ---------------------------------------------------------------------------
__global__ __launch_bounds__(256) void enc_init_kernel(const float* __restrict__ x,
                                                       const float* __restrict__ fch_w,
                                                       const float* __restrict__ fch_b,
                                                       const float* __restrict__ fcc_w,
                                                       const float* __restrict__ fcc_b,
                                                       _Float16* __restrict__ h0,
                                                       float* __restrict__ c0,
                                                       _Float16* __restrict__ queue) {
    int gid = blockIdx.x * blockDim.x + threadIdx.x;
    int b = gid >> 10;
    int t = gid & (T_ENC - 1);
    float xv = x[(size_t)b * T_ENC + t];
    for (int r = 0; r < RCH; ++r) {
        float h = tanhf(fch_w[r] * xv + fch_b[r]);
        float c = tanhf(fcc_w[r] * xv + fcc_b[r]);
        size_t off = ((size_t)b * T_ENC + t) * RCH + r;
        h0[off] = (_Float16)h;
        c0[off] = c;
        if (t >= T_ENC - 128)
            queue[(((size_t)0 * BATCH + b) * QLEN + (t - (T_ENC - 128))) * RCH + r] = (_Float16)h;
    }
}

// ---------------------------------------------------------------------------
// Encoder layer l.  Dynamic LDS: [0,64K) W_l fragments (TDM-staged),
// [64K,96K) z tiles.  Grid (T/32, B); 8 wave32 per workgroup.
// ---------------------------------------------------------------------------
__global__ __launch_bounds__(256) void enc_layer_kernel(int l, int dil,
                                                        const _Float16* __restrict__ wsw,
                                                        const float* __restrict__ bias_all,
                                                        const _Float16* __restrict__ hin,
                                                        _Float16* __restrict__ hout,
                                                        float* __restrict__ cbuf,
                                                        _Float16* __restrict__ queue) {
    extern __shared__ char smem[];
    _Float16* w_lds = (_Float16*)smem;
    float*    z     = (float*)(smem + WLAYER * 2);

    const int b  = blockIdx.y;
    const int t0 = blockIdx.x * 32;
    const int tid  = threadIdx.x;
    const int wv   = tid >> 5;
    const int lane = tid & 31;
    const int half = lane >> 4;
    const int idx  = lane & 15;

    const _Float16* wbase = wsw + (size_t)l * WLAYER;
#if USE_TDM
    if (wv == 0) tdm_load_block(wbase, (unsigned)(uintptr_t)w_lds, (unsigned)WLAYER);
#else
    for (int i = tid; i < WLAYER / 8; i += 256)
        ((uint4*)w_lds)[i] = ((const uint4*)wbase)[i];
#endif
    __builtin_prefetch(hin + ((size_t)b * T_ENC + t0) * RCH, 0, 0);  // global_prefetch_b8
    __syncthreads();

    for (int i = 0; i < 4; ++i) {
        int tile = wv * 4 + i;
        int mt = tile >> 1;
        int nt = tile & 1;
        v8f acc = {};
        for (int kt = 0; kt < 4; ++kt) {
            v16h a = *(const v16h*)(w_lds + ((mt * 4 + kt) * 32 + lane) * 16);
            int t   = t0 + nt * 16 + idx;
            int tt  = (kt >> 1) ? t : (t - dil);     // tap1 = current, tap0 = past
            int ch0 = (kt & 1) * 32 + half * 8;
            v16h bf;
            if (tt >= 0) {
                const _Float16* p = hin + ((size_t)b * T_ENC + tt) * RCH + ch0;
                v8h lo = *(const v8h*)p;
                v8h hi = *(const v8h*)(p + 16);
                for (int j2 = 0; j2 < 8; ++j2) { bf[j2] = lo[j2]; bf[8 + j2] = hi[j2]; }
            } else {
                for (int j2 = 0; j2 < 16; ++j2) bf[j2] = (_Float16)0.0f;
            }
            acc = __builtin_amdgcn_wmma_f32_16x16x32_f16(false, a, false, bf,
                                                         (short)0, acc, false, false);
        }
        for (int v = 0; v < 8; ++v)
            z[(mt * 16 + half * 8 + v) * 32 + nt * 16 + idx] = acc[v];
    }
    __syncthreads();

    const float* bias = bias_all + (size_t)l * 256;
    for (int i = 0; i < 8; ++i) {
        int cell = tid * 8 + i;
        int r  = cell >> 5;
        int tl = cell & 31;
        int t  = t0 + tl;
        float ig = z[(r)       * 32 + tl] + bias[r];
        float cf = z[(64 + r)  * 32 + tl] + bias[64 + r];
        float cg = z[(128 + r) * 32 + tl] + bias[128 + r];
        float eg = z[(192 + r) * 32 + tl] + bias[192 + r];
        size_t off = ((size_t)b * T_ENC + t) * RCH + r;
        float c_new = sigm(ig) * cbuf[off] + tanhf(cf) * sigm(cg);
        float h     = sigm(eg) * tanhf(c_new);
        cbuf[off] = c_new;
        hout[off] = (_Float16)h;
        if (t >= T_ENC - 128)
            queue[(((size_t)(l + 1) * BATCH + b) * QLEN + (t - (T_ENC - 128))) * RCH + r] =
                (_Float16)h;
    }
}

// ---------------------------------------------------------------------------
// Decoder: persistent single workgroup (8 wave32), 64 autoregressive steps.
// Dynamic LDS: W 64K | z 32K | h 4K | c 8K | x 128B.
// ---------------------------------------------------------------------------
__global__ __launch_bounds__(256) void decoder_kernel(const float* __restrict__ enc_x,
                                                      const _Float16* __restrict__ wdec,
                                                      const float* __restrict__ dec_bias,
                                                      const float* __restrict__ dfch_w,
                                                      const float* __restrict__ dfch_b,
                                                      const float* __restrict__ dfcc_w,
                                                      const float* __restrict__ dfcc_b,
                                                      const _Float16* __restrict__ w1,
                                                      const float* __restrict__ b1,
                                                      const float* __restrict__ w2,
                                                      const float* __restrict__ b2,
                                                      _Float16* __restrict__ queue,
                                                      _Float16* __restrict__ skipbuf,
                                                      float* __restrict__ out) {
    extern __shared__ char smem[];
    _Float16* w_lds = (_Float16*)smem;                       // 64 KB
    float*    z     = (float*)(smem + WLAYER * 2);           // 32 KB
    _Float16* h_s   = (_Float16*)(smem + WLAYER * 2 + 32768);// 4 KB
    float*    c_s   = (float*)(smem + WLAYER * 2 + 32768 + 4096); // 8 KB
    float*    x_s   = (float*)(smem + WLAYER * 2 + 32768 + 4096 + 8192);

    const int tid  = threadIdx.x;
    const int wv   = tid >> 5;
    const int lane = tid & 31;
    const int half = lane >> 4;
    const int idx  = lane & 15;

    if (tid < BATCH) x_s[tid] = enc_x[(size_t)tid * T_ENC + (T_ENC - 1)];
    __syncthreads();

#pragma unroll 1
    for (int t = 0; t < DECL; ++t) {
        for (int i = 0; i < 8; ++i) {
            int cell = tid * 8 + i;
            int b = cell >> 6;
            int r = cell & 63;
            float xv = x_s[b];
            h_s[b * RCH + r] = (_Float16)tanhf(dfch_w[r] * xv + dfch_b[r]);
            c_s[b * RCH + r] = tanhf(dfcc_w[r] * xv + dfcc_b[r]);
        }
        __syncthreads();

#pragma unroll 1
        for (int l = 0; l < NLAY; ++l) {
            int dil = 1 << (l & 7);
            // stage W_l to LDS (TDM) + queue[l][b][128+t] = pre-update h
#if USE_TDM
            if (wv == 0)
                tdm_load_block(wdec + (size_t)l * WLAYER, (unsigned)(uintptr_t)w_lds,
                               (unsigned)WLAYER);
#else
            for (int i = tid; i < WLAYER / 8; i += 256)
                ((uint4*)w_lds)[i] = ((const uint4*)(wdec + (size_t)l * WLAYER))[i];
#endif
            for (int i = 0; i < 8; ++i) {
                int cell = tid * 8 + i;
                int b = cell >> 6;
                int r = cell & 63;
                queue[(((size_t)l * BATCH + b) * QLEN + 128 + t) * RCH + r] = h_s[b * RCH + r];
            }
            __syncthreads();

            // z = Wdec_l * [prev ; h]   (M=256, K=128, N=32)
            for (int i = 0; i < 4; ++i) {
                int tile = wv * 4 + i;
                int mt = tile >> 1;
                int nt = tile & 1;
                v8f acc = {};
                for (int kt = 0; kt < 4; ++kt) {
                    v16h a = *(const v16h*)(w_lds + ((mt * 4 + kt) * 32 + lane) * 16);
                    int n   = nt * 16 + idx;
                    int ch0 = (kt & 1) * 32 + half * 8;
                    v16h bf;
                    if (kt >> 1) {
                        const _Float16* p = h_s + n * RCH + ch0;
                        v8h lo = *(const v8h*)p;
                        v8h hi = *(const v8h*)(p + 16);
                        for (int j2 = 0; j2 < 8; ++j2) { bf[j2] = lo[j2]; bf[8 + j2] = hi[j2]; }
                    } else {
                        int u = 128 + t - dil;
                        const _Float16* p =
                            queue + (((size_t)l * BATCH + n) * QLEN + u) * RCH + ch0;
                        v8h lo = *(const v8h*)p;
                        v8h hi = *(const v8h*)(p + 16);
                        for (int j2 = 0; j2 < 8; ++j2) { bf[j2] = lo[j2]; bf[8 + j2] = hi[j2]; }
                    }
                    acc = __builtin_amdgcn_wmma_f32_16x16x32_f16(false, a, false, bf,
                                                                 (short)0, acc, false, false);
                }
                for (int v = 0; v < 8; ++v)
                    z[(mt * 16 + half * 8 + v) * 32 + nt * 16 + idx] = acc[v];
            }
            __syncthreads();

            const float* bias = dec_bias + (size_t)l * 256;
            for (int i = 0; i < 8; ++i) {
                int cell = tid * 8 + i;
                int b = cell >> 6;
                int r = cell & 63;
                float ig = z[(r)       * 32 + b] + bias[r];
                float cf = z[(64 + r)  * 32 + b] + bias[64 + r];
                float cg = z[(128 + r) * 32 + b] + bias[128 + r];
                float eg = z[(192 + r) * 32 + b] + bias[192 + r];
                float c_new = sigm(ig) * c_s[b * RCH + r] + tanhf(cf) * sigm(cg);
                float h     = sigm(eg) * tanhf(c_new);
                c_s[b * RCH + r] = c_new;
                h_s[b * RCH + r] = (_Float16)h;
                skipbuf[((size_t)b * NLAY + l) * RCH + r] = (_Float16)h;
            }
            __syncthreads();
        }

        // y = relu(out1_w @ s + b1)   (M=256, K=1536, N=32)
        for (int i = 0; i < 4; ++i) {
            int tile = wv * 4 + i;
            int mt = tile >> 1;
            int nt = tile & 1;
            v8f acc = {};
            for (int kt = 0; kt < 48; ++kt) {
                v16h a = *(const v16h*)(w1 + (((size_t)mt * 48 + kt) * 32 + lane) * 16);
                int n  = nt * 16 + idx;
                int k0 = kt * 32 + half * 8;
                const _Float16* p = skipbuf + (size_t)n * SKIPK + k0;
                v8h lo = *(const v8h*)p;
                v8h hi = *(const v8h*)(p + 16);
                v16h bf;
                for (int j2 = 0; j2 < 8; ++j2) { bf[j2] = lo[j2]; bf[8 + j2] = hi[j2]; }
                acc = __builtin_amdgcn_wmma_f32_16x16x32_f16(false, a, false, bf,
                                                             (short)0, acc, false, false);
            }
            for (int v = 0; v < 8; ++v) {
                int m = mt * 16 + half * 8 + v;
                float y = acc[v] + b1[m];
                z[m * 32 + nt * 16 + idx] = y > 0.0f ? y : 0.0f;
            }
        }
        __syncthreads();

        if (tid < BATCH) {
            float acc2 = b2[0];
            for (int k = 0; k < HID; ++k) acc2 += z[k * 32 + tid] * w2[k];
            out[(size_t)tid * DECL + t] = acc2;
            x_s[tid] = acc2;
        }
        __syncthreads();
    }
}

// ---------------------------------------------------------------------------
// Host-side orchestration
// ---------------------------------------------------------------------------
extern "C" void kernel_launch(void* const* d_in, const int* in_sizes, int n_in,
                              void* d_out, int out_size, void* d_ws, size_t ws_size,
                              hipStream_t stream) {
    (void)in_sizes; (void)n_in; (void)out_size; (void)ws_size;

    const float* enc_x     = (const float*)d_in[0];
    const float* enc_fch_w = (const float*)d_in[2];
    const float* enc_fch_b = (const float*)d_in[3];
    const float* enc_fcc_w = (const float*)d_in[4];
    const float* enc_fcc_b = (const float*)d_in[5];
    const float* enc_cnn_w = (const float*)d_in[6];
    const float* enc_cnn_b = (const float*)d_in[7];
    const float* dec_fch_w = (const float*)d_in[8];
    const float* dec_fch_b = (const float*)d_in[9];
    const float* dec_fcc_w = (const float*)d_in[10];
    const float* dec_fcc_b = (const float*)d_in[11];
    const float* dec_cnn_w = (const float*)d_in[12];
    const float* dec_cnn_b = (const float*)d_in[13];
    const float* out1_w    = (const float*)d_in[14];
    const float* out1_b    = (const float*)d_in[15];
    const float* out2_w    = (const float*)d_in[16];
    const float* out2_b    = (const float*)d_in[17];
    float* out = (float*)d_out;

    char* ws = (char*)d_ws;
    size_t off = 0;
    auto take = [&](size_t bytes) -> char* {
        char* p = ws + off;
        off = (off + bytes + 255) & ~(size_t)255;
        return p;
    };
    const size_t CNNW_ELEMS = (size_t)NLAY * WLAYER;
    const size_t W1_ELEMS   = (size_t)16 * 48 * 32 * 16;
    _Float16* wenc  = (_Float16*)take(CNNW_ELEMS * 2);
    _Float16* wdec  = (_Float16*)take(CNNW_ELEMS * 2);
    _Float16* w1    = (_Float16*)take(W1_ELEMS * 2);
    _Float16* h_a   = (_Float16*)take((size_t)BATCH * T_ENC * RCH * 2);
    _Float16* h_b   = (_Float16*)take((size_t)BATCH * T_ENC * RCH * 2);
    float*    cbuf  = (float*)   take((size_t)BATCH * T_ENC * RCH * 4);
    _Float16* queue = (_Float16*)take((size_t)NLAY * BATCH * QLEN * RCH * 2);
    _Float16* skipb = (_Float16*)take((size_t)BATCH * SKIPK * 2);

    swizzle_cnn_w_kernel<<<(CNNW_ELEMS + 255) / 256, 256, 0, stream>>>(enc_cnn_w, wenc);
    swizzle_cnn_w_kernel<<<(CNNW_ELEMS + 255) / 256, 256, 0, stream>>>(dec_cnn_w, wdec);
    swizzle_out1_w_kernel<<<(W1_ELEMS + 255) / 256, 256, 0, stream>>>(out1_w, w1);

    enc_init_kernel<<<(BATCH * T_ENC) / 256, 256, 0, stream>>>(
        enc_x, enc_fch_w, enc_fch_b, enc_fcc_w, enc_fcc_b, h_a, cbuf, queue);

    const size_t ENC_SMEM = (size_t)WLAYER * 2 + 32768;                    // 96 KB
    for (int l = 0; l < NLAY - 1; ++l) {
        _Float16* hin  = (l & 1) ? h_b : h_a;
        _Float16* hout = (l & 1) ? h_a : h_b;
        enc_layer_kernel<<<dim3(T_ENC / 32, BATCH), 256, ENC_SMEM, stream>>>(
            l, 1 << (l & 7), wenc, enc_cnn_b, hin, hout, cbuf, queue);
    }

    const size_t DEC_SMEM = (size_t)WLAYER * 2 + 32768 + 4096 + 8192 + 256; // ~108.5 KB
    decoder_kernel<<<1, 256, DEC_SMEM, stream>>>(
        enc_x, wdec, dec_cnn_b, dec_fch_w, dec_fch_b, dec_fcc_w, dec_fcc_b,
        w1, out1_b, out2_w, out2_b, queue, skipb, out);
}
